// MEAM_50929722196886
// MI455X (gfx1250) — compile-verified, matrix operand order checked by reading, not run on hardware
//
#include <hip/hip_runtime.h>
#include <math.h>

// ---------------------------------------------------------------------------
// MEAM forward for MI455X (gfx1250).
// Dense GEMM-shaped ops via v_wmma_f32_16x16x32_f16, wave32, one wave per
// 16(M)x64(N) tile, 4 independent accumulators, b128 fragment loads, and a
// software-pipelined (double-buffered) K loop so loads for step k+1 are in
// flight while step k's WMMAs execute.
// Weights pre-transposed/copied to f16 [Npad,K] (zero padded) so inner-loop
// B loads are unconditional.
// ---------------------------------------------------------------------------

typedef __attribute__((ext_vector_type(16))) _Float16 v16h;
typedef __attribute__((ext_vector_type(8)))  _Float16 v8h;
typedef __attribute__((ext_vector_type(8)))  float    v8f;

#define NB   32
#define NF   83
#define NS   10
#define NMF  9
#define NP   40
#define NPP  80
#define ND   128

__device__ __forceinline__ float gelu_exact(float x) {
  return 0.5f * x * (1.0f + erff(x * 0.70710678118f));
}

__device__ __forceinline__ v16h cat8(v8h lo, v8h hi) {
  return __builtin_shufflevector(lo, hi, 0, 1, 2, 3, 4, 5, 6, 7,
                                 8, 9, 10, 11, 12, 13, 14, 15);
}

__device__ __forceinline__ v16h frag_from_f32(float4 a0, float4 a1,
                                              float4 a2, float4 a3) {
  v16h a;
  a[0]  = (_Float16)a0.x; a[1]  = (_Float16)a0.y;
  a[2]  = (_Float16)a0.z; a[3]  = (_Float16)a0.w;
  a[4]  = (_Float16)a1.x; a[5]  = (_Float16)a1.y;
  a[6]  = (_Float16)a1.z; a[7]  = (_Float16)a1.w;
  a[8]  = (_Float16)a2.x; a[9]  = (_Float16)a2.y;
  a[10] = (_Float16)a2.z; a[11] = (_Float16)a2.w;
  a[12] = (_Float16)a3.x; a[13] = (_Float16)a3.y;
  a[14] = (_Float16)a3.z; a[15] = (_Float16)a3.w;
  return a;
}

// ---------------------------------------------------------------------------
// Weight pre-passes. Npad is N rounded up to 64; rows n>=N are zero-filled.
// ---------------------------------------------------------------------------
extern "C" __global__ void
meam_transpose_f16(const float* __restrict__ W, _Float16* __restrict__ WT,
                   int K, int N, int Npad)
{
  int idx = blockIdx.x * blockDim.x + threadIdx.x;
  if (idx >= Npad * K) return;
  int n = idx / K, k = idx - n * K;
  WT[idx] = (n < N) ? (_Float16)W[(size_t)k * N + n] : (_Float16)0.f;
}

extern "C" __global__ void
meam_pad_f16(const float* __restrict__ W, _Float16* __restrict__ WT,
             int K, int N, int Npad)
{
  int idx = blockIdx.x * blockDim.x + threadIdx.x;
  if (idx >= Npad * K) return;
  int n = idx / K;
  WT[idx] = (n < N) ? (_Float16)W[idx] : (_Float16)0.f;
}

// ---------------------------------------------------------------------------
// GEMM fragment bundle (raw loads; converted at compute time).
// ---------------------------------------------------------------------------
struct GFrag {
  float4 a0, a1, a2, a3;
  v8h b0l, b0h, b1l, b1h, b2l, b2h, b3l, b3h;
};

__device__ __forceinline__ void
g_load(GFrag& f, const float* ap, const _Float16* b0, const _Float16* b1,
       const _Float16* b2, const _Float16* b3)
{
  f.a0  = *(const float4*)(ap);
  f.a1  = *(const float4*)(ap + 4);
  f.a2  = *(const float4*)(ap + 16);
  f.a3  = *(const float4*)(ap + 20);
  f.b0l = *(const v8h*)(b0); f.b0h = *(const v8h*)(b0 + 16);
  f.b1l = *(const v8h*)(b1); f.b1h = *(const v8h*)(b1 + 16);
  f.b2l = *(const v8h*)(b2); f.b2h = *(const v8h*)(b2 + 16);
  f.b3l = *(const v8h*)(b3); f.b3h = *(const v8h*)(b3 + 16);
}

__device__ __forceinline__ void
g_compute(const GFrag& f, v8f& acc0, v8f& acc1, v8f& acc2, v8f& acc3)
{
  v16h a  = frag_from_f32(f.a0, f.a1, f.a2, f.a3);
  v16h b0 = cat8(f.b0l, f.b0h);
  v16h b1 = cat8(f.b1l, f.b1h);
  v16h b2 = cat8(f.b2l, f.b2h);
  v16h b3 = cat8(f.b3l, f.b3h);
  acc0 = __builtin_amdgcn_wmma_f32_16x16x32_f16(false, a, false, b0,
                                                (short)0, acc0, false, false);
  acc1 = __builtin_amdgcn_wmma_f32_16x16x32_f16(false, a, false, b1,
                                                (short)0, acc1, false, false);
  acc2 = __builtin_amdgcn_wmma_f32_16x16x32_f16(false, a, false, b2,
                                                (short)0, acc2, false, false);
  acc3 = __builtin_amdgcn_wmma_f32_16x16x32_f16(false, a, false, b3,
                                                (short)0, acc3, false, false);
}

// ---------------------------------------------------------------------------
// GEMM: C[M,N] = act(A[M,K]*B + bias). A f32 row-major, BT f16 [Npad,K].
// Requires M%16==0, K%32==0, BT padded to ceil(N/64)*64 rows.
// Grid: (ceil(N/64), M/16), 32 threads (one wave).
// ---------------------------------------------------------------------------
extern "C" __global__ void __launch_bounds__(32)
meam_gemm_wmma(const float* __restrict__ A, const _Float16* __restrict__ BT,
               const float* __restrict__ bias, float* __restrict__ C,
               int M, int N, int K, int act, int hasBias)
{
  int lane = threadIdx.x;
  int tn = blockIdx.x, tm = blockIdx.y;
  int nl   = lane & 15;
  int off  = (lane >> 4) << 3;      // 0 or 8: per-lane K-run offset
  int m    = tm * 16 + nl;
  const float* arow = A + (size_t)m * K + off;

  int n0 = tn * 64 + nl;
  const _Float16* brow0 = BT + (size_t)(n0)      * K + off;
  const _Float16* brow1 = BT + (size_t)(n0 + 16) * K + off;
  const _Float16* brow2 = BT + (size_t)(n0 + 32) * K + off;
  const _Float16* brow3 = BT + (size_t)(n0 + 48) * K + off;

  v8f acc0 = {}, acc1 = {}, acc2 = {}, acc3 = {};

  GFrag cur;
  g_load(cur, arow, brow0, brow1, brow2, brow3);
  int k0 = 0;
#pragma unroll 2
  for (; k0 + 32 < K; k0 += 32) {
    GFrag nxt;
    int kn = k0 + 32;
    g_load(nxt, arow + kn, brow0 + kn, brow1 + kn, brow2 + kn, brow3 + kn);
    __builtin_prefetch(arow + kn + 256, 0, 0);
    g_compute(cur, acc0, acc1, acc2, acc3);
    cur = nxt;
  }
  g_compute(cur, acc0, acc1, acc2, acc3);

  int mbase = tm * 16 + ((lane >> 4) << 3);
#pragma unroll
  for (int nt = 0; nt < 4; ++nt) {
    int ncol = n0 + nt * 16;
    if (ncol >= N) continue;
    v8f acc = (nt == 0) ? acc0 : (nt == 1) ? acc1 : (nt == 2) ? acc2 : acc3;
    float bia = hasBias ? bias[ncol] : 0.f;
#pragma unroll
    for (int r = 0; r < 8; ++r) {
      float v = acc[r] + bia;
      if (act == 1)      v = fmaxf(v, 0.f);
      else if (act == 2) v = gelu_exact(v);
      C[(size_t)(mbase + r) * N + ncol] = v;
    }
  }
}

// ---------------------------------------------------------------------------
// Conv fragment bundle.
// ---------------------------------------------------------------------------
struct CFrag {
  float av[16];
  v8h b0l, b0h, b1l, b1h, b2l, b2h, b3l, b3h;
};

__device__ __forceinline__ void
c_load(CFrag& f, int k0, int off, int ihb, int iwb, int IH, int IW,
       const float* xb, const _Float16* b0, const _Float16* b1,
       const _Float16* b2, const _Float16* b3)
{
#pragma unroll
  for (int e = 0; e < 16; ++e) {
    int kg = k0 + off + ((e < 8) ? e : (8 + e));   // runs: +0..7, +16..23
    int c  = kg / 9;
    int t  = kg - 9 * c;
    int kh = t / 3, kw = t - 3 * (t / 3);
    int ih = ihb + kh, iw = iwb + kw;
    float v = 0.f;
    if (ih >= 0 && ih < IH && iw >= 0 && iw < IW)
      v = xb[((size_t)c * IH + ih) * IW + iw];
    f.av[e] = v;
  }
  f.b0l = *(const v8h*)(b0 + k0); f.b0h = *(const v8h*)(b0 + k0 + 16);
  f.b1l = *(const v8h*)(b1 + k0); f.b1h = *(const v8h*)(b1 + k0 + 16);
  f.b2l = *(const v8h*)(b2 + k0); f.b2h = *(const v8h*)(b2 + k0 + 16);
  f.b3l = *(const v8h*)(b3 + k0); f.b3h = *(const v8h*)(b3 + k0 + 16);
}

__device__ __forceinline__ void
c_compute(const CFrag& f, v8f& acc0, v8f& acc1, v8f& acc2, v8f& acc3)
{
  v16h a;
#pragma unroll
  for (int e = 0; e < 16; ++e) a[e] = (_Float16)f.av[e];
  v16h b0 = cat8(f.b0l, f.b0h);
  v16h b1 = cat8(f.b1l, f.b1h);
  v16h b2 = cat8(f.b2l, f.b2h);
  v16h b3 = cat8(f.b3l, f.b3h);
  acc0 = __builtin_amdgcn_wmma_f32_16x16x32_f16(false, a, false, b0,
                                                (short)0, acc0, false, false);
  acc1 = __builtin_amdgcn_wmma_f32_16x16x32_f16(false, a, false, b1,
                                                (short)0, acc1, false, false);
  acc2 = __builtin_amdgcn_wmma_f32_16x16x32_f16(false, a, false, b2,
                                                (short)0, acc2, false, false);
  acc3 = __builtin_amdgcn_wmma_f32_16x16x32_f16(false, a, false, b3,
                                                (short)0, acc3, false, false);
}

// ---------------------------------------------------------------------------
// Implicit-GEMM 3x3 stride-2 pad-1 conv (NCHW), f16 WMMA.
// M=B*OH*OW (mult of 16), K=IC*9 (mult of 32), N=OC<=64.
// WT is f16 [64][K], rows >= OC zero-filled. Grid: (1, M/16).
// ---------------------------------------------------------------------------
extern "C" __global__ void __launch_bounds__(32)
meam_conv3x3s2_wmma(const float* __restrict__ X, const _Float16* __restrict__ WT,
                    const float* __restrict__ bias, float* __restrict__ Y,
                    int Bn, int IC, int IH, int IW, int OC, int OH, int OW,
                    int act)
{
  int lane = threadIdx.x;
  int tm = blockIdx.y;
  int nl  = lane & 15;
  int off = (lane >> 4) << 3;
  int K = IC * 9;

  int p = tm * 16 + nl;
  int bIdx = p / (OH * OW);
  int prem = p - bIdx * OH * OW;
  int oh = prem / OW, ow = prem - (prem / OW) * OW;
  int ihb = oh * 2 - 1, iwb = ow * 2 - 1;
  const float* xb = X + (size_t)bIdx * IC * IH * IW;

  const _Float16* brow0 = WT + (size_t)(nl)      * K + off;
  const _Float16* brow1 = WT + (size_t)(nl + 16) * K + off;
  const _Float16* brow2 = WT + (size_t)(nl + 32) * K + off;
  const _Float16* brow3 = WT + (size_t)(nl + 48) * K + off;

  v8f acc0 = {}, acc1 = {}, acc2 = {}, acc3 = {};

  CFrag cur;
  c_load(cur, 0, off, ihb, iwb, IH, IW, xb, brow0, brow1, brow2, brow3);
  int k0 = 0;
#pragma unroll 2
  for (; k0 + 32 < K; k0 += 32) {
    CFrag nxt;
    c_load(nxt, k0 + 32, off, ihb, iwb, IH, IW, xb, brow0, brow1, brow2, brow3);
    c_compute(cur, acc0, acc1, acc2, acc3);
    cur = nxt;
  }
  c_compute(cur, acc0, acc1, acc2, acc3);

  int mbase = tm * 16 + ((lane >> 4) << 3);
#pragma unroll
  for (int nt = 0; nt < 4; ++nt) {
    int oc = nt * 16 + nl;
    if (oc >= OC) continue;
    v8f acc = (nt == 0) ? acc0 : (nt == 1) ? acc1 : (nt == 2) ? acc2 : acc3;
    float bia = bias[oc];
#pragma unroll
    for (int r = 0; r < 8; ++r) {
      int mm = mbase + r;
      int b2 = mm / (OH * OW);
      int rm = mm - b2 * OH * OW;
      int oh2 = rm / OW, ow2 = rm - (rm / OW) * OW;
      float v = acc[r] + bia;
      if (act == 2) v = gelu_exact(v);
      Y[(((size_t)b2 * OC + oc) * OH + oh2) * OW + ow2] = v;
    }
  }
}

// ---------------------------------------------------------------------------
// LayerNorm over channel dim (NCHW) + exact GELU, in place. eps = 1e-6.
// ---------------------------------------------------------------------------
extern "C" __global__ void
meam_ln2d_gelu(float* __restrict__ X, const float* __restrict__ w,
               const float* __restrict__ bg, int Bn, int Cc, int HW)
{
  int idx = blockIdx.x * blockDim.x + threadIdx.x;
  if (idx >= Bn * HW) return;
  int b = idx / HW, hw = idx - b * HW;
  float* px = X + (size_t)b * Cc * HW + hw;
  float mu = 0.f;
  for (int c = 0; c < Cc; ++c) mu += px[(size_t)c * HW];
  mu /= (float)Cc;
  float var = 0.f;
  for (int c = 0; c < Cc; ++c) { float d = px[(size_t)c * HW] - mu; var += d * d; }
  var /= (float)Cc;
  float inv = rsqrtf(var + 1e-6f);
  for (int c = 0; c < Cc; ++c) {
    float v = (px[(size_t)c * HW] - mu) * inv * w[c] + bg[c];
    px[(size_t)c * HW] = gelu_exact(v);
  }
}

// ---------------------------------------------------------------------------
// Ragged segment gather + zero pad: vfeat[B,F,128] -> padded[B,S,MF,128].
// ---------------------------------------------------------------------------
extern "C" __global__ void
meam_pad_segments(const float* __restrict__ vfeat, const int* __restrict__ ind,
                  float* __restrict__ padded)
{
  int b = blockIdx.x, t = threadIdx.x;
  __shared__ int pos[NF];
  if (t == 0) {
    int prev = -1, start = 0;
    for (int f = 0; f < NF; ++f) {
      int s = ind[b * NF + f];
      if (s != prev) { start = f; prev = s; }
      pos[f] = f - start;
    }
  }
  float* pb = padded + (size_t)b * NS * NMF * ND;
  for (int i = t; i < NS * NMF * ND; i += 128) pb[i] = 0.f;
  __syncthreads();
  for (int f = 0; f < NF; ++f) {
    int s = ind[b * NF + f];
    pb[((size_t)s * NMF + pos[f]) * ND + t] =
        vfeat[((size_t)b * NF + f) * ND + t];
  }
}

// ---------------------------------------------------------------------------
// Row-wise attention (4 heads x 64): qkv packed [G,n,768] (q|k|v of 256).
// grid=(n, 4, G), block=96. out: [G,n,256] head-concat.
// ---------------------------------------------------------------------------
extern "C" __global__ void
meam_attn_rows(const float* __restrict__ qkv, float* __restrict__ out, int n)
{
  int i = blockIdx.x, h = blockIdx.y, g = blockIdx.z;
  int t = threadIdx.x;
  const float* base = qkv + (size_t)g * n * 768;
  __shared__ float sc[96];
  if (t < n) {
    const float* qi = base + (size_t)i * 768 + h * 64;
    const float* kt = base + (size_t)t * 768 + 256 + h * 64;
    float s = 0.f;
    for (int d = 0; d < 64; ++d) s += qi[d] * kt[d];
    sc[t] = s * 0.125f;   // 64^-0.5
  }
  __syncthreads();
  if (t == 0) {
    float mx = -3.4e38f;
    for (int j = 0; j < n; ++j) mx = fmaxf(mx, sc[j]);
    float den = 0.f;
    for (int j = 0; j < n; ++j) { sc[j] = __expf(sc[j] - mx); den += sc[j]; }
    float r = 1.f / den;
    for (int j = 0; j < n; ++j) sc[j] *= r;
  }
  __syncthreads();
  if (t < 64) {
    float o = 0.f;
    for (int j = 0; j < n; ++j)
      o += sc[j] * base[(size_t)j * 768 + 512 + h * 64 + t];
    out[((size_t)g * n + i) * 256 + h * 64 + t] = o;
  }
}

extern "C" __global__ void
meam_reduce_mean(const float* __restrict__ in, float* __restrict__ out,
                 int n, float inv, int ostride, int ooff)
{
  int g = blockIdx.x, d = threadIdx.x;
  float s = 0.f;
  for (int i = 0; i < n; ++i) s += in[((size_t)g * n + i) * ND + d];
  out[(size_t)g * ostride + ooff + d] = s * inv;
}

extern "C" __global__ void
meam_build_v_temporal(const float* __restrict__ vseg, float* __restrict__ vt)
{
  int b = blockIdx.x, row = blockIdx.y, d = threadIdx.x;
  float v = 0.f;
  if (row < NS) {
    float fr = __expf(-logf(1.0e4f) * (float)(d & 63) / 64.f);
    float ang = (float)row * fr;
    v = vseg[((size_t)b * NS + row) * ND + d] +
        ((d < 64) ? __sinf(ang) : __cosf(ang));
  }
  vt[((size_t)b * NF + row) * ND + d] = v;
}

extern "C" __global__ void
meam_build_t_temporal(const float* __restrict__ tfeat, const float* __restrict__ treg,
                      const float* __restrict__ fps, const float* __restrict__ tot,
                      float* __restrict__ tt)
{
  int b = blockIdx.x, p = blockIdx.y, d = threadIdx.x;
  float v = 0.f;
  if (p < NP) {
    float dur = treg[((size_t)b * NP + p) * 2 + 1] - treg[((size_t)b * NP + p) * 2 + 0];
    float L = logf(1.0e4f);
    float dv;
    if (d < 64) {
      float val = dur / fps[b];
      float fr = __expf(-L * (float)(d & 31) / 32.f);
      dv = (d < 32) ? __sinf(val * fr) : __cosf(val * fr);
    } else {
      int dd = d - 64;
      float val = dur / tot[b];
      float fr = __expf(-L * (float)(dd & 31) / 32.f);
      dv = (dd < 32) ? __sinf(val * fr) : __cosf(val * fr);
    }
    float fr2 = __expf(-L * (float)(d & 63) / 64.f);
    float pe = (d < 64) ? __sinf((float)p * fr2) : __cosf((float)p * fr2);
    v = tfeat[((size_t)b * NPP + p) * ND + d] + dv + pe;
  }
  tt[((size_t)b * NPP + p) * ND + d] = v;
}

// ---------------------------------------------------------------------------
// Transformer layer self-attention (2 tokens, 2 heads) + residual + LN1.
// ---------------------------------------------------------------------------
extern "C" __global__ void
meam_trm_mha_ln(const float* __restrict__ xin, const float* __restrict__ in_w,
                const float* __restrict__ in_b, const float* __restrict__ ow,
                const float* __restrict__ ob, const float* __restrict__ lnw,
                const float* __restrict__ lnb, float* __restrict__ x1)
{
  int b = blockIdx.x, d = threadIdx.x;
  __shared__ float x[2][ND], q[2][ND], kk[2][ND], vv[2][ND], oo[2][ND];
  __shared__ float att[2][2][2];
  __shared__ float red[2];

  x[0][d] = xin[b * 256 + d];
  x[1][d] = xin[b * 256 + ND + d];
  __syncthreads();

  for (int t2 = 0; t2 < 2; ++t2) {
    float sq = in_b[d], sk = in_b[128 + d], sv = in_b[256 + d];
    for (int k2 = 0; k2 < ND; ++k2) {
      float xv = x[t2][k2];
      sq += xv * in_w[d * ND + k2];
      sk += xv * in_w[(128 + d) * ND + k2];
      sv += xv * in_w[(256 + d) * ND + k2];
    }
    q[t2][d] = sq; kk[t2][d] = sk; vv[t2][d] = sv;
  }
  __syncthreads();

  if (d < 8) {
    int h = d >> 2, t2 = (d >> 1) & 1, j = d & 1;
    float s = 0.f;
    for (int e = 0; e < 64; ++e) s += q[t2][h * 64 + e] * kk[j][h * 64 + e];
    att[h][t2][j] = s * 0.125f;
  }
  __syncthreads();
  if (d < 4) {
    int h = d >> 1, t2 = d & 1;
    float a0 = att[h][t2][0], a1 = att[h][t2][1];
    float m = fmaxf(a0, a1);
    float e0 = __expf(a0 - m), e1 = __expf(a1 - m);
    float r = 1.f / (e0 + e1);
    att[h][t2][0] = e0 * r; att[h][t2][1] = e1 * r;
  }
  __syncthreads();
  int h = d >> 6;
  float o0 = att[h][0][0] * vv[0][d] + att[h][0][1] * vv[1][d];
  float o1 = att[h][1][0] * vv[0][d] + att[h][1][1] * vv[1][d];
  __syncthreads();
  oo[0][d] = o0; oo[1][d] = o1;
  __syncthreads();

  for (int t2 = 0; t2 < 2; ++t2) {
    float s = ob[d];
    for (int k2 = 0; k2 < ND; ++k2) s += oo[t2][k2] * ow[d * ND + k2];
    float val = x[t2][d] + s;
    q[t2][d] = val;
    __syncthreads();
    if (d == 0) {
      float mu = 0.f;
      for (int k2 = 0; k2 < ND; ++k2) mu += q[t2][k2];
      mu /= (float)ND;
      float va = 0.f;
      for (int k2 = 0; k2 < ND; ++k2) { float dd = q[t2][k2] - mu; va += dd * dd; }
      va /= (float)ND;
      red[0] = mu; red[1] = rsqrtf(va + 1e-5f);
    }
    __syncthreads();
    x1[b * 256 + t2 * ND + d] = (val - red[0]) * red[1] * lnw[d] + lnb[d];
    __syncthreads();
  }
}

extern "C" __global__ void
meam_trm_final(const float* __restrict__ x1, const float* __restrict__ ff,
               const float* __restrict__ lnw, const float* __restrict__ lnb,
               const float* __restrict__ layout, float* __restrict__ meam)
{
  int b = blockIdx.x, d = threadIdx.x;
  __shared__ float buf[2][ND];
  __shared__ float red[2];
  float acc = 0.f;
  for (int t2 = 0; t2 < 2; ++t2) {
    float val = x1[b * 256 + t2 * ND + d] + ff[b * 256 + t2 * ND + d];
    buf[t2][d] = val;
    __syncthreads();
    if (d == 0) {
      float mu = 0.f;
      for (int k2 = 0; k2 < ND; ++k2) mu += buf[t2][k2];
      mu /= (float)ND;
      float va = 0.f;
      for (int k2 = 0; k2 < ND; ++k2) { float dd = buf[t2][k2] - mu; va += dd * dd; }
      va /= (float)ND;
      red[0] = mu; red[1] = rsqrtf(va + 1e-5f);
    }
    __syncthreads();
    acc += (val - red[0]) * red[1] * lnw[d] + lnb[d];
    __syncthreads();
  }
  meam[b * 256 + ND + d] = acc * 0.5f;
  meam[b * 256 + d] = layout[b * ND + d];
}

// ---------------------------------------------------------------------------
// Host orchestration
// ---------------------------------------------------------------------------
static inline dim3 gemm_grid(int M, int N) {
  return dim3((N + 63) / 64, M / 16);
}
static inline int pad64(int n) { return (n + 63) & ~63; }

extern "C" void kernel_launch(void* const* d_in, const int* in_sizes, int n_in,
                              void* d_out, int out_size, void* d_ws, size_t ws_size,
                              hipStream_t stream) {
  (void)in_sizes; (void)n_in; (void)out_size; (void)ws_size;

  const float* ocr_pattern = (const float*)d_in[0];   // [32,256,64,64]
  const float* ocr_phrase  = (const float*)d_in[1];   // [32,80,512]
  const float* ocr_time    = (const float*)d_in[2];   // [32,40,2]
  const float* vis_frames  = (const float*)d_in[3];   // [32,83,512]
  const int*   seg_ind     = (const int*)d_in[4];     // [32,83]
  const float* fps         = (const float*)d_in[6];   // [32]
  const float* totf        = (const float*)d_in[7];   // [32]

  const float* P_[38];
  for (int i = 0; i < 38; ++i) P_[i] = (const float*)d_in[8 + i];
  const float *conv1_w = P_[0],  *conv1_b = P_[1];
  const float *ln2d_w  = P_[2],  *ln2d_b  = P_[3];
  const float *conv2_w = P_[4],  *conv2_b = P_[5];
  const float *mlp1_w  = P_[6],  *mlp1_b  = P_[7];
  const float *mlp2_w  = P_[8],  *mlp2_b  = P_[9];
  const float *mlp3_w  = P_[10], *mlp3_b  = P_[11];
  const float *proj_t_w = P_[12], *proj_t_b = P_[13];
  const float *proj_v_w = P_[14], *proj_v_b = P_[15];
  const float *intra_qkv = P_[16], *intra_out = P_[17];
  const float *t_qkv = P_[18], *t_out = P_[19];
  const float *v_qkv = P_[20], *v_out = P_[21];
  const float *trm_in_w = P_[22], *trm_in_b = P_[23];
  const float *trm_out_w = P_[24], *trm_out_b = P_[25];
  const float *trm_ln1_w = P_[26], *trm_ln1_b = P_[27];
  const float *trm_ln2_w = P_[28], *trm_ln2_b = P_[29];
  const float *trm_ff1_w = P_[30], *trm_ff1_b = P_[31];
  const float *trm_ff2_w = P_[32], *trm_ff2_b = P_[33];
  const float *cls1_w = P_[34], *cls1_b = P_[35];
  const float *cls2_w = P_[36], *cls2_b = P_[37];

  float* out = (float*)d_out;  // [32,2]

  // Byte-granular workspace carve-up, 256B aligned blocks.
  char* base = (char*)d_ws;
  size_t off = 0;
  auto allocf = [&](size_t n) {
    float* p = (float*)(base + off); off += ((n * 4 + 255) & ~(size_t)255); return p;
  };
  auto alloch = [&](size_t n) {
    _Float16* p = (_Float16*)(base + off); off += ((n * 2 + 255) & ~(size_t)255); return p;
  };

  float* h1     = allocf((size_t)NB * 64 * 32 * 32);
  float* h2     = allocf((size_t)NB * 16 * 16 * 16);
  float* m1     = allocf((size_t)NB * 2048);
  float* m2     = allocf((size_t)NB * 512);
  float* layout = allocf((size_t)NB * ND);
  float* vfeat  = allocf((size_t)NB * NF * ND);
  float* padded = allocf((size_t)NB * NS * NMF * ND);
  float* iqkv   = allocf((size_t)NB * NS * NMF * 768);
  float* iatt   = allocf((size_t)NB * NS * NMF * 256);
  float* iproj  = allocf((size_t)NB * NS * NMF * ND);
  float* vseg   = allocf((size_t)NB * NS * ND);
  float* v_temp = allocf((size_t)NB * NF * ND);
  float* tfeat  = allocf((size_t)NB * NPP * ND);
  float* t_temp = allocf((size_t)NB * NPP * ND);
  float* tqkv   = allocf((size_t)NB * NPP * 768);
  float* vqkv   = allocf((size_t)NB * NF * 768);
  float* tatt   = allocf((size_t)NB * NPP * 256);
  float* vatt   = allocf((size_t)NB * NF * 256);
  float* tproj  = allocf((size_t)NB * NPP * ND);
  float* vproj  = allocf((size_t)NB * NF * ND);
  float* xbuf   = allocf((size_t)NB * 2 * ND);
  float* x1     = allocf((size_t)NB * 2 * ND);
  float* ffh    = allocf((size_t)NB * 2 * 2048);
  float* ffb    = allocf((size_t)NB * 2 * ND);
  float* meam   = allocf((size_t)NB * 256);
  float* c1     = allocf((size_t)NB * ND);

  // f16 [Npad,K] weights (transposed or pad-copied), zero rows for n>=N.
  _Float16* T_mlp1  = alloch((size_t)2048 * 4096);
  _Float16* T_mlp2  = alloch((size_t)512 * 2048);
  _Float16* T_mlp3  = alloch((size_t)128 * 512);
  _Float16* T_pt    = alloch((size_t)128 * 512);
  _Float16* T_pv    = alloch((size_t)128 * 512);
  _Float16* T_iqkv  = alloch((size_t)768 * 128);
  _Float16* T_iout  = alloch((size_t)128 * 256);
  _Float16* T_tqkv  = alloch((size_t)768 * 128);
  _Float16* T_tout  = alloch((size_t)128 * 256);
  _Float16* T_vqkv  = alloch((size_t)768 * 128);
  _Float16* T_vout  = alloch((size_t)128 * 256);
  _Float16* T_ff1   = alloch((size_t)2048 * 128);
  _Float16* T_ff2   = alloch((size_t)128 * 2048);
  _Float16* T_cls1  = alloch((size_t)128 * 256);
  _Float16* T_cls2  = alloch((size_t)64 * 128);     // Npad=64
  _Float16* T_conv1 = alloch((size_t)64 * 2304);
  _Float16* T_conv2 = alloch((size_t)64 * 576);     // Npad=64 (OC=16)

  auto tr = [&](const float* W, _Float16* WT, int K, int N) {
    int np = pad64(N);
    int tot = np * K;
    meam_transpose_f16<<<(tot + 255) / 256, 256, 0, stream>>>(W, WT, K, N, np);
  };
  tr(mlp1_w, T_mlp1, 4096, 2048);
  tr(mlp2_w, T_mlp2, 2048, 512);
  tr(mlp3_w, T_mlp3, 512, 128);
  tr(proj_t_w, T_pt, 512, 128);
  tr(proj_v_w, T_pv, 512, 128);
  tr(intra_qkv, T_iqkv, 128, 768);
  tr(intra_out, T_iout, 256, 128);
  tr(t_qkv, T_tqkv, 128, 768);
  tr(t_out, T_tout, 256, 128);
  tr(v_qkv, T_vqkv, 128, 768);
  tr(v_out, T_vout, 256, 128);
  tr(trm_ff1_w, T_ff1, 128, 2048);
  tr(trm_ff2_w, T_ff2, 2048, 128);
  tr(cls1_w, T_cls1, 256, 128);
  tr(cls2_w, T_cls2, 128, 2);
  meam_pad_f16<<<(64 * 2304 + 255) / 256, 256, 0, stream>>>(conv1_w, T_conv1, 2304, 64, 64);
  meam_pad_f16<<<(64 * 576 + 255) / 256, 256, 0, stream>>>(conv2_w, T_conv2, 576, 16, 64);

  // ---------------- OCR layout-pattern branch ----------------
  meam_conv3x3s2_wmma<<<dim3(1, 2048), 32, 0, stream>>>(
      ocr_pattern, T_conv1, conv1_b, h1, NB, 256, 64, 64, 64, 32, 32, 0);
  meam_ln2d_gelu<<<(NB * 1024 + 255) / 256, 256, 0, stream>>>(
      h1, ln2d_w, ln2d_b, NB, 64, 1024);
  meam_conv3x3s2_wmma<<<dim3(1, 512), 32, 0, stream>>>(
      h1, T_conv2, conv2_b, h2, NB, 64, 32, 32, 16, 16, 16, 2);
  meam_gemm_wmma<<<gemm_grid(NB, 2048), 32, 0, stream>>>(
      h2, T_mlp1, mlp1_b, m1, NB, 2048, 4096, 1, 1);
  meam_gemm_wmma<<<gemm_grid(NB, 512), 32, 0, stream>>>(
      m1, T_mlp2, mlp2_b, m2, NB, 512, 2048, 1, 1);
  meam_gemm_wmma<<<gemm_grid(NB, ND), 32, 0, stream>>>(
      m2, T_mlp3, mlp3_b, layout, NB, ND, 512, 1, 1);

  // ---------------- Visual branch ----------------
  meam_gemm_wmma<<<gemm_grid(NB * NF, ND), 32, 0, stream>>>(
      vis_frames, T_pv, proj_v_b, vfeat, NB * NF, ND, 512, 0, 1);
  meam_pad_segments<<<NB, 128, 0, stream>>>(vfeat, seg_ind, padded);
  meam_gemm_wmma<<<gemm_grid(NB * NS * NMF, 768), 32, 0, stream>>>(
      padded, T_iqkv, nullptr, iqkv, NB * NS * NMF, 768, ND, 0, 0);
  meam_attn_rows<<<dim3(NMF, 4, NB * NS), 96, 0, stream>>>(iqkv, iatt, NMF);
  meam_gemm_wmma<<<gemm_grid(NB * NS * NMF, ND), 32, 0, stream>>>(
      iatt, T_iout, nullptr, iproj, NB * NS * NMF, ND, 256, 0, 0);
  meam_reduce_mean<<<NB * NS, 128, 0, stream>>>(iproj, vseg, NMF, 1.f / NMF, ND, 0);
  meam_build_v_temporal<<<dim3(NB, NF), 128, 0, stream>>>(vseg, v_temp);

  // ---------------- OCR phrase branch ----------------
  meam_gemm_wmma<<<gemm_grid(NB * NPP, ND), 32, 0, stream>>>(
      ocr_phrase, T_pt, proj_t_b, tfeat, NB * NPP, ND, 512, 0, 1);
  meam_build_t_temporal<<<dim3(NB, NPP), 128, 0, stream>>>(
      tfeat, ocr_time, fps, totf, t_temp);

  meam_gemm_wmma<<<gemm_grid(NB * NPP, 768), 32, 0, stream>>>(
      t_temp, T_tqkv, nullptr, tqkv, NB * NPP, 768, ND, 0, 0);
  meam_attn_rows<<<dim3(NPP, 4, NB), 96, 0, stream>>>(tqkv, tatt, NPP);
  meam_gemm_wmma<<<gemm_grid(NB * NPP, ND), 32, 0, stream>>>(
      tatt, T_tout, nullptr, tproj, NB * NPP, ND, 256, 0, 0);
  meam_reduce_mean<<<NB, 128, 0, stream>>>(tproj, xbuf, NPP, 1.f / NP, 256, 0);

  meam_gemm_wmma<<<gemm_grid(NB * NF, 768), 32, 0, stream>>>(
      v_temp, T_vqkv, nullptr, vqkv, NB * NF, 768, ND, 0, 0);
  meam_attn_rows<<<dim3(NF, 4, NB), 96, 0, stream>>>(vqkv, vatt, NF);
  meam_gemm_wmma<<<gemm_grid(NB * NF, ND), 32, 0, stream>>>(
      vatt, T_vout, nullptr, vproj, NB * NF, ND, 256, 0, 0);
  meam_reduce_mean<<<NB, 128, 0, stream>>>(vproj, xbuf, NF, 1.f / NS, 256, ND);

  // ---------------- multimodal transformer layer ----------------
  meam_trm_mha_ln<<<NB, 128, 0, stream>>>(
      xbuf, trm_in_w, trm_in_b, trm_out_w, trm_out_b, trm_ln1_w, trm_ln1_b, x1);
  meam_gemm_wmma<<<gemm_grid(NB * 2, 2048), 32, 0, stream>>>(
      x1, T_ff1, trm_ff1_b, ffh, NB * 2, 2048, ND, 1, 1);
  meam_gemm_wmma<<<gemm_grid(NB * 2, ND), 32, 0, stream>>>(
      ffh, T_ff2, trm_ff2_b, ffb, NB * 2, ND, 2048, 0, 1);
  meam_trm_final<<<NB, 128, 0, stream>>>(
      x1, ffb, trm_ln2_w, trm_ln2_b, layout, meam);

  // ---------------- classifier ----------------
  meam_gemm_wmma<<<gemm_grid(NB, ND), 32, 0, stream>>>(
      meam, T_cls1, cls1_b, c1, NB, ND, 256, 1, 1);
  meam_gemm_wmma<<<gemm_grid(NB, 2), 32, 0, stream>>>(
      c1, T_cls2, cls2_b, out, NB, 2, ND, 0, 1);
}
